// MultiHeadSelfAttention_14568529068534
// MI455X (gfx1250) — matrix operand order repaired and checked by reading, not compiled, not run on hardware
//
#include <hip/hip_runtime.h>

// ---------------------------------------------------------------------------
// Fused multi-head self-attention forward for MI455X (gfx1250, wave32).
//   B=4, S=2048, D_MODEL=1024, H=16, DEPTH=64
// Pipeline (all bf16 WMMA with f32 accumulate):
//   1) q,k,v f32 -> bf16            (Xq,Xk,Xv)
//   2) wq,wk,wv,wo f32 -> bf16, transposed to [N][K]  (Wqt,Wkt,Wvt,Wot)
//   3) Q/K/V projections: 8192x1024x1024 WMMA GEMM + bias -> bf16
//      (register double-buffered K-loop: loads for k0+32 issued before the
//       wmmas for k0)
//   4) V -> per-head transposed [B,H,64,S]
//   5) flash attention: K/V tiles staged cooperatively into double-buffered
//      LDS via global_load_async_to_lds_b128 (ASYNCcnt); all K fragments
//      loaded up-front, V fragments loaded before the softmax block so DS
//      latency hides under the exp/reduction VALU work
//   6) output projection WMMA GEMM + bias -> f32 d_out
// Workspace: 3*16MB (X*) + 4*2MB (W*t) + 3*16MB (Q/K/V) = 104MB; Vt and Oc
// alias Xq/Xk which are dead by then.
// ---------------------------------------------------------------------------

#define B_  4
#define S_  2048
#define DM  1024
#define H_  16
#define DH  64

typedef __attribute__((ext_vector_type(16))) __bf16 bf16x16;
typedef __attribute__((ext_vector_type(8)))  float  f32x8;

union FragAB { bf16x16 v; unsigned int u[8]; };

__device__ __forceinline__ unsigned short f2bf(float f) {
  union { float f; unsigned int u; } x; x.f = f;
  unsigned int r = x.u + 0x7FFFu + ((x.u >> 16) & 1u);   // round-to-nearest-even
  return (unsigned short)(r >> 16);
}

// reductions across a 16-lane half of the wave (rows of a C-fragment half)
__device__ __forceinline__ float half_max(float v) {
  v = fmaxf(v, __shfl_xor(v, 8, 32));
  v = fmaxf(v, __shfl_xor(v, 4, 32));
  v = fmaxf(v, __shfl_xor(v, 2, 32));
  v = fmaxf(v, __shfl_xor(v, 1, 32));
  return v;
}
__device__ __forceinline__ float half_sum(float v) {
  v += __shfl_xor(v, 8, 32);
  v += __shfl_xor(v, 4, 32);
  v += __shfl_xor(v, 2, 32);
  v += __shfl_xor(v, 1, 32);
  return v;
}

// async copy: 16 bytes global -> LDS, tracked by ASYNCcnt (CDNA5 path).
// GV addressing: vdst = LDS byte address, v[pair] = 64-bit global address.
__device__ __forceinline__ void async_copy_b128(const void* gsrc, void* ldst) {
  unsigned lds = (unsigned)(uintptr_t)ldst;   // low 32 bits = LDS offset
  asm volatile("global_load_async_to_lds_b128 %0, %1, off"
               :: "v"(lds), "v"(gsrc)
               : "memory");
}
__device__ __forceinline__ void wait_async0() {
  asm volatile("s_wait_asynccnt 0x0" ::: "memory");
}

// ---------------- conversion / layout kernels ------------------------------

__global__ void cvt_f32_bf16(const float* __restrict__ x,
                             unsigned short* __restrict__ y, int n) {
  int i = blockIdx.x * blockDim.x + threadIdx.x;
  if (i < n) y[i] = f2bf(x[i]);
}

// W [K x N] f32 row-major -> Wt [N x K] bf16 (so B-fragments load contiguous)
__global__ void cvt_transpose_w(const float* __restrict__ W,
                                unsigned short* __restrict__ Wt) {
  int i = blockIdx.x * blockDim.x + threadIdx.x;   // flat over N*K = 1M
  int k = i & (DM - 1);
  int n = i >> 10;
  Wt[i] = f2bf(W[(size_t)k * DM + n]);
}

// Vb [B,S,DM] bf16 -> Vt [B,H,DH,S] bf16 (coalesced writes)
__global__ void transpose_v(const unsigned short* __restrict__ Vb,
                            unsigned short* __restrict__ Vt) {
  int i = blockIdx.x * blockDim.x + threadIdx.x;   // flat over 8M
  int s = i & (S_ - 1);
  int d = (i >> 11) & (DH - 1);
  int h = (i >> 17) & (H_ - 1);
  int b = i >> 21;
  Vt[i] = Vb[((size_t)(b * S_ + s)) * DM + h * DH + d];
}

// ---------------- WMMA GEMM:  out = A @ W + bias ---------------------------
// A  : [M x K] bf16 row-major
// Bt : [N x K] bf16 (W transposed) -> B[k][n] = Bt[n*K+k]
// Wave computes a 16(M) x 64(N) strip; block of 8 waves = 64 x 128 tile.
// Register double-buffered over k0 so global loads overlap the wmmas.
template <int OUT_BF16>
__global__ __launch_bounds__(256) void gemm_bf16_wmma(
    const unsigned short* __restrict__ A,
    const unsigned short* __restrict__ Bt,
    const float* __restrict__ bias,
    void* __restrict__ out, int M, int N, int K) {
  const int lane = threadIdx.x & 31;
  const int wave = threadIdx.x >> 5;
  const int wm = wave & 3, wn = wave >> 2;
  const int mbase = blockIdx.y * 64 + wm * 16;
  const int nbase = blockIdx.x * 128 + wn * 64;
  const int half = lane >> 4;
  const int l16  = lane & 15;
  const int klo  = half << 4;

  const unsigned short* Arow = A + (size_t)(mbase + l16) * K;

  auto loadA = [&](FragAB& fr, int k0) {
#pragma unroll
    for (int vv = 0; vv < 8; ++vv) {
      int kk = klo + 2 * vv;
      fr.u[vv] = *(const unsigned int*)(Arow + k0 + kk);
    }
  };
  auto loadB = [&](FragAB& fr, int t, int k0) {
    const unsigned short* Brow =
        Bt + (size_t)(nbase + t * 16 + l16) * K + k0 + klo;
#pragma unroll
    for (int vv = 0; vv < 8; ++vv)
      fr.u[vv] = *(const unsigned int*)(Brow + 2 * vv);
  };

  f32x8 c[4] = {};
  FragAB a[2];
  FragAB bb[2][4];

  loadA(a[0], 0);
#pragma unroll
  for (int t = 0; t < 4; ++t) loadB(bb[0][t], t, 0);

  for (int k0 = 0; k0 < K; k0 += 32) {
    int cur = (k0 >> 5) & 1;
    int nxt = cur ^ 1;
    if (k0 + 32 < K) {
      __builtin_prefetch(Arow + k0 + 128, 0, 1);   // global_prefetch_b8
      loadA(a[nxt], k0 + 32);
#pragma unroll
      for (int t = 0; t < 4; ++t) loadB(bb[nxt][t], t, k0 + 32);
    }
#pragma unroll
    for (int t = 0; t < 4; ++t)
      c[t] = __builtin_amdgcn_wmma_f32_16x16x32_bf16(
          false, a[cur].v, false, bb[cur][t].v, (short)0, c[t], false, false);
  }

#pragma unroll
  for (int t = 0; t < 4; ++t) {
    int col = nbase + t * 16 + l16;
    float bbia = bias[col];
#pragma unroll
    for (int r = 0; r < 8; ++r) {
      int row = mbase + r + 8 * half;
      float val = c[t][r] + bbia;
      if (OUT_BF16)
        ((unsigned short*)out)[(size_t)row * N + col] = f2bf(val);
      else
        ((float*)out)[(size_t)row * N + col] = val;
    }
  }
}

// ---------------- flash attention ------------------------------------------
// Block = 8 waves = 128 query rows of one (b,h); each wave owns 16 rows.
// K/V tiles for 32 keys are staged cooperatively into double-buffered LDS via
// async copies (tile i+1 staged while computing tile i). Per 32-key step:
// all 4 K fragments loaded, 4 wmma for S=QK^T, V fragments loaded (latency
// hidden under softmax), online softmax, P restaged via per-wave LDS tile
// (C-layout -> A-layout), 4 wmma for O += P V.
__global__ __launch_bounds__(256) void flash_attn_wmma(
    const unsigned short* __restrict__ Q,   // [B,S,DM]
    const unsigned short* __restrict__ Kb,  // [B,S,DM]
    const unsigned short* __restrict__ Vt,  // [B,H,DH,S]
    unsigned short* __restrict__ Oc) {      // [B,S,DM] (head-concat)
  // padded rows keep every async B128 dest 16B-aligned and spread LDS banks
  __shared__ unsigned short kt[2][32][72];  // [buf][key][depth], 144B rows
  __shared__ unsigned short vt[2][64][40];  // [buf][depth][key],  80B rows
  __shared__ unsigned int pbuf32[8][256];   // per-wave 16x32 bf16 P tile

  const int tid  = threadIdx.x;
  const int lane = tid & 31;
  const int wave = tid >> 5;
  const int half = lane >> 4;
  const int l16  = lane & 15;
  const int klo  = half << 4;
  const int b = blockIdx.z, h = blockIdx.y;
  const int qbase = blockIdx.x * 128 + wave * 16;

  unsigned short* pb = (unsigned short*)pbuf32[wave];

  const unsigned short* Kbase = Kb + ((size_t)(b * S_)) * DM + h * DH;
  const unsigned short* Vbase = Vt + ((size_t)((b * H_ + h) * DH)) * S_;

  // staging pattern: 256 threads move K tile (32 rows x 128B) and V tile
  // (64 rows x 64B) as one B128 chunk each per thread.
  const int krow = tid >> 3, kch = tid & 7;   // K: chunk = 16B of a 128B row
  const int vrow = tid >> 2, vch = tid & 3;   // V: chunk = 16B of a  64B row
  auto stage = [&](int buf, int kv) {
    async_copy_b128(Kbase + (size_t)(kv + krow) * DM + kch * 8,
                    &kt[buf][krow][kch * 8]);
    async_copy_b128(Vbase + (size_t)vrow * S_ + kv + vch * 8,
                    &vt[buf][vrow][vch * 8]);
  };

  // Preload Q A-fragments for depth 0-31 and 32-63.
  const unsigned short* Qrow = Q + ((size_t)(b * S_ + qbase + l16)) * DM + h * DH;
  FragAB aq0, aq1;
#pragma unroll
  for (int vv = 0; vv < 8; ++vv) {
    int kk = klo + 2 * vv;
    aq0.u[vv] = *(const unsigned int*)(Qrow + kk);
    aq1.u[vv] = *(const unsigned int*)(Qrow + 32 + kk);
  }

  float mrow[8], lrow[8];
#pragma unroll
  for (int r = 0; r < 8; ++r) { mrow[r] = -1e30f; lrow[r] = 0.0f; }
  f32x8 o[4] = {};

  const float scale = 0.125f;   // 1/sqrt(64)

  stage(0, 0);   // prologue: fill buffer 0

  for (int kv = 0; kv < S_; kv += 32) {
    const int cur = (kv >> 5) & 1;

    // my staged async copies done; barrier also guarantees every wave has
    // finished reading the buffer that the next stage() will overwrite.
    wait_async0();
    __syncthreads();
    if (kv + 32 < S_) stage(cur ^ 1, kv + 32);

    // ---- load all 4 K B-fragments up-front ------------------------------
    // bk[st][ks]: st = key subtile (0:keys 0-15, 1:keys 16-31),
    //             ks = depth step (0: d 0-31, 1: d 32-63)
    FragAB bk[2][2];
#pragma unroll
    for (int st = 0; st < 2; ++st) {
      const unsigned short* Krow = &kt[cur][st * 16 + l16][klo];
#pragma unroll
      for (int vv = 0; vv < 8; ++vv) {
        bk[st][0].u[vv] = *(const unsigned int*)(Krow + 2 * vv);
        bk[st][1].u[vv] = *(const unsigned int*)(Krow + 32 + 2 * vv);
      }
    }

    // ---- S = Q K^T for 32 keys: two 16x16 tiles -------------------------
    f32x8 s0 = {}, s1 = {};
    s0 = __builtin_amdgcn_wmma_f32_16x16x32_bf16(
        false, aq0.v, false, bk[0][0].v, (short)0, s0, false, false);
    s1 = __builtin_amdgcn_wmma_f32_16x16x32_bf16(
        false, aq0.v, false, bk[1][0].v, (short)0, s1, false, false);
    s0 = __builtin_amdgcn_wmma_f32_16x16x32_bf16(
        false, aq1.v, false, bk[0][1].v, (short)0, s0, false, false);
    s1 = __builtin_amdgcn_wmma_f32_16x16x32_bf16(
        false, aq1.v, false, bk[1][1].v, (short)0, s1, false, false);

    // ---- issue V B-fragment loads now; latency hides under softmax ------
    FragAB bvf[4];
#pragma unroll
    for (int t = 0; t < 4; ++t) {
      const unsigned short* Vrow = &vt[cur][t * 16 + l16][klo];
#pragma unroll
      for (int vv = 0; vv < 8; ++vv)
        bvf[t].u[vv] = *(const unsigned int*)(Vrow + 2 * vv);
    }

    // ---- online softmax over the 32 new columns -------------------------
    float alpha[8];
#pragma unroll
    for (int r = 0; r < 8; ++r) {
      float v0 = s0[r] * scale, v1 = s1[r] * scale;
      float tmax = half_max(fmaxf(v0, v1));
      float mnew = fmaxf(mrow[r], tmax);
      float al   = __expf(mrow[r] - mnew);
      float p0   = __expf(v0 - mnew);
      float p1   = __expf(v1 - mnew);
      lrow[r] = lrow[r] * al + half_sum(p0 + p1);
      mrow[r] = mnew;
      alpha[r] = al;
      // stage P into per-wave LDS: row = r + 8*half, cols l16 / 16+l16
      int rowoff = (r + 8 * half) * 32;
      pb[rowoff + l16]      = f2bf(p0);
      pb[rowoff + 16 + l16] = f2bf(p1);
    }
#pragma unroll
    for (int t = 0; t < 4; ++t)
#pragma unroll
      for (int r = 0; r < 8; ++r) o[t][r] *= alpha[r];

    // ensure the cross-lane LDS writes have landed before A-layout reads
    asm volatile("s_wait_dscnt 0x0" ::: "memory");

    // ---- reload P as a 16x32 A-fragment ---------------------------------
    FragAB pa;
#pragma unroll
    for (int vv = 0; vv < 8; ++vv) {
      int kk = klo + 2 * vv;
      pa.u[vv] = *(const unsigned int*)(pb + l16 * 32 + kk);
    }

    // ---- O += P @ V over 4 depth tiles ----------------------------------
#pragma unroll
    for (int t = 0; t < 4; ++t)
      o[t] = __builtin_amdgcn_wmma_f32_16x16x32_bf16(
          false, pa.v, false, bvf[t].v, (short)0, o[t], false, false);
  }

  // ---- normalize and store head-concatenated output ---------------------
#pragma unroll
  for (int t = 0; t < 4; ++t)
#pragma unroll
    for (int r = 0; r < 8; ++r) {
      float val = o[t][r] / lrow[r];
      Oc[((size_t)(b * S_ + qbase + r + 8 * half)) * DM + h * DH + t * 16 + l16] =
          f2bf(val);
    }
}

// ---------------------------------------------------------------------------

extern "C" void kernel_launch(void* const* d_in, const int* in_sizes, int n_in,
                              void* d_out, int out_size, void* d_ws, size_t ws_size,
                              hipStream_t stream) {
  (void)in_sizes; (void)n_in; (void)out_size; (void)ws_size;

  const float* v  = (const float*)d_in[0];
  const float* k  = (const float*)d_in[1];
  const float* q  = (const float*)d_in[2];
  const float* wq = (const float*)d_in[3];
  const float* bq = (const float*)d_in[4];
  const float* wk = (const float*)d_in[5];
  const float* bk = (const float*)d_in[6];
  const float* wv = (const float*)d_in[7];
  const float* bv = (const float*)d_in[8];
  const float* wo = (const float*)d_in[9];
  const float* bo = (const float*)d_in[10];

  const int    NTOK  = B_ * S_;                 // 8192
  const int    NELEM = NTOK * DM;               // 8M
  const size_t ACT_B = (size_t)NELEM * 2;       // 16MB per bf16 activation
  const size_t W_B   = (size_t)DM * DM * 2;     // 2MB per bf16 weight

  char* ws = (char*)d_ws;
  unsigned short* Xq  = (unsigned short*)(ws);               // 16MB
  unsigned short* Xk  = (unsigned short*)(ws + ACT_B);       // 16MB
  unsigned short* Xv  = (unsigned short*)(ws + 2 * ACT_B);   // 16MB
  unsigned short* Wqt = (unsigned short*)(ws + 3 * ACT_B);
  unsigned short* Wkt = (unsigned short*)(ws + 3 * ACT_B + W_B);
  unsigned short* Wvt = (unsigned short*)(ws + 3 * ACT_B + 2 * W_B);
  unsigned short* Wot = (unsigned short*)(ws + 3 * ACT_B + 3 * W_B);
  unsigned short* Qb  = (unsigned short*)(ws + 3 * ACT_B + 4 * W_B);
  unsigned short* Kbf = (unsigned short*)(ws + 4 * ACT_B + 4 * W_B);
  unsigned short* Vb  = (unsigned short*)(ws + 5 * ACT_B + 4 * W_B);
  // X buffers are dead after the projections: reuse for Vt / O-concat.
  unsigned short* Vt  = Xq;
  unsigned short* Oc  = Xk;

  const int T = 256;
  const int cvtBlocks = (NELEM + T - 1) / T;        // 32768
  const int wBlocks   = (DM * DM) / T;              // 4096

  // 1) activations -> bf16
  cvt_f32_bf16<<<cvtBlocks, T, 0, stream>>>(q, Xq, NELEM);
  cvt_f32_bf16<<<cvtBlocks, T, 0, stream>>>(k, Xk, NELEM);
  cvt_f32_bf16<<<cvtBlocks, T, 0, stream>>>(v, Xv, NELEM);

  // 2) weights -> bf16 transposed [N][K]
  cvt_transpose_w<<<wBlocks, T, 0, stream>>>(wq, Wqt);
  cvt_transpose_w<<<wBlocks, T, 0, stream>>>(wk, Wkt);
  cvt_transpose_w<<<wBlocks, T, 0, stream>>>(wv, Wvt);
  cvt_transpose_w<<<wBlocks, T, 0, stream>>>(wo, Wot);

  // 3) projections (bias fused), bf16 out
  dim3 gGemm(DM / 128, NTOK / 64);                  // (8, 128)
  gemm_bf16_wmma<1><<<gGemm, T, 0, stream>>>(Xq, Wqt, bq, Qb,  NTOK, DM, DM);
  gemm_bf16_wmma<1><<<gGemm, T, 0, stream>>>(Xk, Wkt, bk, Kbf, NTOK, DM, DM);
  gemm_bf16_wmma<1><<<gGemm, T, 0, stream>>>(Xv, Wvt, bv, Vb,  NTOK, DM, DM);

  // 4) per-head V transpose [B,H,64,S]
  transpose_v<<<cvtBlocks, T, 0, stream>>>(Vb, Vt);

  // 5) flash attention -> head-concat bf16
  dim3 gAttn(S_ / 128, H_, B_);                     // (16, 16, 4)
  flash_attn_wmma<<<gAttn, T, 0, stream>>>(Qb, Kbf, Vt, Oc);

  // 6) output projection, f32 out
  gemm_bf16_wmma<0><<<gGemm, T, 0, stream>>>(Oc, Wot, bo, d_out, NTOK, DM, DM);
}
